// MOELayer_57621281243505
// MI455X (gfx1250) — matrix operand order repaired and checked by reading, not compiled, not run on hardware
//
#include <hip/hip_runtime.h>
#include <hip/hip_bf16.h>
#include <math.h>

// ---------------- problem constants ----------------
constexpr int D_MODEL = 768;
constexpr int D_HID   = 3072;
constexpr int D_OUT   = 768;
constexpr int N_EXP   = 8;
constexpr int T_TOK   = 2 * 2048;          // B*S tokens
constexpr int NSLOT   = T_TOK * 2;         // token-expert pairs (top-2)
constexpr float LN_EPS = 1e-5f;

// GEMM tiling: block tile 128x64x32, 8 waves, each wave 32x32 (4 WMMA frags)
constexpr int TM = 128, TN = 64, TK = 32;
constexpr int MAX_MT = (NSLOT + TM - 1) / TM;   // 64 tile slots per expert (upper bound)
constexpr int SLOT_PAD = NSLOT + TM;            // pad so last-tile A reads stay in bounds

typedef __attribute__((ext_vector_type(16))) _Float16 v16h;
typedef __attribute__((ext_vector_type(8)))  _Float16 v8h;
typedef __attribute__((ext_vector_type(8)))  float    v8f;

__device__ __forceinline__ float gelu_tanh(float v) {
    float v3 = v * v * v;
    return 0.5f * v * (1.0f + tanhf(0.7978845608028654f * (v + 0.044715f * v3)));
}

__device__ __forceinline__ v16h cat8(v8h lo, v8h hi) {
    return __builtin_shufflevector(lo, hi, 0,1,2,3,4,5,6,7,8,9,10,11,12,13,14,15);
}

// ---------------- 1) zero per-expert meta ----------------
__global__ void moe_zero(int* counts, int* cursor) {
    int i = threadIdx.x;
    if (i < N_EXP) { counts[i] = 0; cursor[i] = 0; }
}

// ---------------- 2) router: logits, softmax, top-2 ----------------
__global__ __launch_bounds__(128) void moe_router(
    const float* __restrict__ x, const float* __restrict__ Wr,
    const float* __restrict__ br,
    int* __restrict__ tok_idx, float* __restrict__ tok_w, int* __restrict__ counts)
{
    const int t   = blockIdx.x;
    const int tid = threadIdx.x;
    __shared__ float red[128 * N_EXP];

    float acc[N_EXP];
    #pragma unroll
    for (int e = 0; e < N_EXP; ++e) acc[e] = 0.0f;

    const float* xr = x + (size_t)t * D_MODEL;
    for (int d = tid; d < D_MODEL; d += 128) {
        float xv = xr[d];
        const float4* w = reinterpret_cast<const float4*>(Wr + (size_t)d * N_EXP);
        float4 a = w[0], b = w[1];
        acc[0] += xv * a.x; acc[1] += xv * a.y; acc[2] += xv * a.z; acc[3] += xv * a.w;
        acc[4] += xv * b.x; acc[5] += xv * b.y; acc[6] += xv * b.z; acc[7] += xv * b.w;
    }
    #pragma unroll
    for (int e = 0; e < N_EXP; ++e) red[tid * N_EXP + e] = acc[e];
    __syncthreads();
    for (int s = 64; s > 0; s >>= 1) {
        if (tid < s) {
            #pragma unroll
            for (int e = 0; e < N_EXP; ++e)
                red[tid * N_EXP + e] += red[(tid + s) * N_EXP + e];
        }
        __syncthreads();
    }
    if (tid == 0) {
        float lg[N_EXP];
        #pragma unroll
        for (int e = 0; e < N_EXP; ++e) lg[e] = red[e] + br[e];
        float m = lg[0];
        #pragma unroll
        for (int e = 1; e < N_EXP; ++e) m = fmaxf(m, lg[e]);
        float p[N_EXP];
        #pragma unroll
        for (int e = 0; e < N_EXP; ++e) p[e] = expf(lg[e] - m);
        int i0 = 0;
        #pragma unroll
        for (int e = 1; e < N_EXP; ++e) if (p[e] > p[i0]) i0 = e;
        int i1 = (i0 == 0) ? 1 : 0;
        #pragma unroll
        for (int e = 0; e < N_EXP; ++e) if (e != i0 && p[e] > p[i1]) i1 = e;
        float denom = p[i0] + p[i1];
        tok_idx[2 * t]     = i0;
        tok_idx[2 * t + 1] = i1;
        tok_w[2 * t]       = p[i0] / denom;
        tok_w[2 * t + 1]   = p[i1] / denom;
        atomicAdd(&counts[i0], 1);
        atomicAdd(&counts[i1], 1);
    }
}

// ---------------- 3) exclusive prefix over 8 experts ----------------
__global__ void moe_offsets(const int* __restrict__ counts, int* __restrict__ offs) {
    if (threadIdx.x == 0) {
        int a = 0;
        for (int e = 0; e < N_EXP; ++e) { offs[e] = a; a += counts[e]; }
        offs[N_EXP] = a;
    }
}

// ---------------- 4) assign each (token,k) pair a slot grouped by expert ----
__global__ __launch_bounds__(256) void moe_assign(
    const int* __restrict__ tok_idx, const int* __restrict__ offs,
    int* __restrict__ cursor, int* __restrict__ row_token, int* __restrict__ slot_of)
{
    int t = blockIdx.x * blockDim.x + threadIdx.x;
    if (t >= T_TOK) return;
    #pragma unroll
    for (int k = 0; k < 2; ++k) {
        int e   = tok_idx[2 * t + k];
        int pos = offs[e] + atomicAdd(&cursor[e], 1);
        row_token[pos]     = t;
        slot_of[2 * t + k] = pos;   // slot permutation only; per-token math order is fixed
    }
}

// ---------------- 5) grouped GEMM1: H = gelu(x @ W1[e] + b1[e])  (f16 out) ---
__global__ __launch_bounds__(256, 1) void moe_gemm1(
    const float* __restrict__ x, const float* __restrict__ W1,
    const float* __restrict__ b1, const int* __restrict__ offs,
    const int* __restrict__ row_token, _Float16* __restrict__ Hh)
{
    const int e    = blockIdx.x >> 6;
    const int mt   = blockIdx.x & (MAX_MT - 1);
    const int base = offs[e];
    const int cnt  = offs[e + 1] - base;
    if (mt * TM >= cnt) return;
    const int n0  = blockIdx.y * TN;
    const int tid = threadIdx.x;

    __shared__ _Float16 As[2][TM * TK];   // 2 x 8KB
    __shared__ _Float16 Bs[2][TN * TK];   // 2 x 4KB
    __shared__ int      toks[TM];

    if (tid < TM) {
        int r = mt * TM + tid;
        int rr = (r < cnt) ? r : (cnt - 1);   // clamp padding rows to a valid row
        toks[tid] = row_token[base + rr];     // (stores for padded rows are guarded)
    }
    __syncthreads();

    const float* W1e = W1 + (size_t)e * D_MODEL * D_HID;

    const int lane = tid & 31;
    const int wv   = tid >> 5;
    const int wm   = wv & 3;     // 4 wave-rows of 32
    const int wn   = wv >> 2;    // 2 wave-cols of 32
    v8f c[2][2] = {{{}, {}}, {{}, {}}};

    const int ai = tid >> 1;           // A row   0..127
    const int ak = (tid & 1) * 16;     // A k     0,16
    const int bk = tid >> 3;           // B k row 0..31
    const int bn = (tid & 7) * 8;      // B n     0..56

    // loop-invariant source pointers (no per-iteration LDS reads / 64-bit mads)
    const int    tok   = toks[ai];
    const float* arow  = x + (size_t)tok * D_MODEL + ak;
    const float* bcol  = W1e + (size_t)bk * D_HID + n0 + bn;

    float4 fa0, fa1, fa2, fa3, fb0, fb1;

    auto fetch = [&](int kb) {   // issue global loads for one stage into regs
        const float4* p = reinterpret_cast<const float4*>(arow + kb);
        fa0 = p[0]; fa1 = p[1]; fa2 = p[2]; fa3 = p[3];
        const float* src = bcol + (size_t)kb * D_HID;
        const float4* q = reinterpret_cast<const float4*>(src);
        fb0 = q[0]; fb1 = q[1];
        __builtin_prefetch(src + (size_t)TK * D_HID, 0, 1);  // stage after next
    };
    auto stash = [&](int buf) {  // fp32 -> f16, regs -> LDS
        v8h h0, h1;
        h0[0]=(_Float16)fa0.x; h0[1]=(_Float16)fa0.y; h0[2]=(_Float16)fa0.z; h0[3]=(_Float16)fa0.w;
        h0[4]=(_Float16)fa1.x; h0[5]=(_Float16)fa1.y; h0[6]=(_Float16)fa1.z; h0[7]=(_Float16)fa1.w;
        h1[0]=(_Float16)fa2.x; h1[1]=(_Float16)fa2.y; h1[2]=(_Float16)fa2.z; h1[3]=(_Float16)fa2.w;
        h1[4]=(_Float16)fa3.x; h1[5]=(_Float16)fa3.y; h1[6]=(_Float16)fa3.z; h1[7]=(_Float16)fa3.w;
        *reinterpret_cast<v8h*>(&As[buf][ai * TK + ak])     = h0;
        *reinterpret_cast<v8h*>(&As[buf][ai * TK + ak + 8]) = h1;
        _Float16* bs = &Bs[buf][0];
        bs[(bn + 0) * TK + bk] = (_Float16)fb0.x;
        bs[(bn + 1) * TK + bk] = (_Float16)fb0.y;
        bs[(bn + 2) * TK + bk] = (_Float16)fb0.z;
        bs[(bn + 3) * TK + bk] = (_Float16)fb0.w;
        bs[(bn + 4) * TK + bk] = (_Float16)fb1.x;
        bs[(bn + 5) * TK + bk] = (_Float16)fb1.y;
        bs[(bn + 6) * TK + bk] = (_Float16)fb1.z;
        bs[(bn + 7) * TK + bk] = (_Float16)fb1.w;
    };
    auto compute = [&](int buf) {  // 2 A-frags x 2 B-frags -> 4 WMMA
        const int koA = (lane >> 4) * 8;           // ISA 16-bit A layout
        const int m0  = wm * 32 + (lane & 15);
        const _Float16* ap0 = &As[buf][m0 * TK];
        const _Float16* ap1 = &As[buf][(m0 + 16) * TK];
        v16h a0 = cat8(*reinterpret_cast<const v8h*>(ap0 + koA),
                       *reinterpret_cast<const v8h*>(ap0 + 16 + koA));
        v16h a1 = cat8(*reinterpret_cast<const v8h*>(ap1 + koA),
                       *reinterpret_cast<const v8h*>(ap1 + 16 + koA));
        const int koB = (lane >> 4) * 8;           // ISA 16-bit B layout
        const int nc  = wn * 32 + (lane & 15);
        const _Float16* bp0 = &Bs[buf][nc * TK + koB];
        const _Float16* bp1 = &Bs[buf][(nc + 16) * TK + koB];
        v16h b0 = cat8(*reinterpret_cast<const v8h*>(bp0),
                       *reinterpret_cast<const v8h*>(bp0 + 16));
        v16h b1f = cat8(*reinterpret_cast<const v8h*>(bp1),
                        *reinterpret_cast<const v8h*>(bp1 + 16));
        c[0][0] = __builtin_amdgcn_wmma_f32_16x16x32_f16(false, a0, false, b0,  (short)0, c[0][0], false, false);
        c[0][1] = __builtin_amdgcn_wmma_f32_16x16x32_f16(false, a0, false, b1f, (short)0, c[0][1], false, false);
        c[1][0] = __builtin_amdgcn_wmma_f32_16x16x32_f16(false, a1, false, b0,  (short)0, c[1][0], false, false);
        c[1][1] = __builtin_amdgcn_wmma_f32_16x16x32_f16(false, a1, false, b1f, (short)0, c[1][1], false, false);
    };

    const int KT = D_MODEL / TK;   // 24
    fetch(0);
    stash(0);
    for (int kt = 0; kt < KT; ++kt) {
        if (kt + 1 < KT) fetch((kt + 1) * TK);   // loads fly under the WMMAs
        __syncthreads();
        compute(kt & 1);
        if (kt + 1 < KT) stash((kt + 1) & 1);
    }

    // epilogue: bias + gelu -> f16 H
    const int mb = (lane >> 4) * 8;
    const int nl = lane & 15;
    const float* b1e = b1 + (size_t)e * D_HID;
    #pragma unroll
    for (int fm = 0; fm < 2; ++fm) {
        #pragma unroll
        for (int fn = 0; fn < 2; ++fn) {
            const int nglob = n0 + wn * 32 + fn * 16 + nl;
            const float bias = b1e[nglob];
            #pragma unroll
            for (int r = 0; r < 8; ++r) {
                int rl = wm * 32 + fm * 16 + mb + r;
                if (mt * TM + rl < cnt) {
                    float v = gelu_tanh(c[fm][fn][r] + bias);
                    Hh[(size_t)(base + mt * TM + rl) * D_HID + nglob] = (_Float16)v;
                }
            }
        }
    }
}

// ---------------- 6) grouped GEMM2: Y = H @ W2[e] + b2[e]  (fp32 out) -------
__global__ __launch_bounds__(256, 1) void moe_gemm2(
    const _Float16* __restrict__ Hh, const float* __restrict__ W2,
    const float* __restrict__ b2, const int* __restrict__ offs,
    float* __restrict__ Y)
{
    const int e    = blockIdx.x >> 6;
    const int mt   = blockIdx.x & (MAX_MT - 1);
    const int base = offs[e];
    const int cnt  = offs[e + 1] - base;
    if (mt * TM >= cnt) return;
    const int n0  = blockIdx.y * TN;
    const int tid = threadIdx.x;

    __shared__ _Float16 As[2][TM * TK];
    __shared__ _Float16 Bs[2][TN * TK];

    const float* W2e = W2 + (size_t)e * D_HID * D_OUT;

    const int lane = tid & 31;
    const int wv   = tid >> 5;
    const int wm   = wv & 3;
    const int wn   = wv >> 2;
    v8f c[2][2] = {{{}, {}}, {{}, {}}};

    const int ai = tid >> 1;
    const int ak = (tid & 1) * 16;
    const int bk = tid >> 3;
    const int bn = (tid & 7) * 8;

    // loop-invariant source pointers
    const _Float16* arow = Hh + (size_t)(base + mt * TM + ai) * D_HID + ak;  // padded rows in bounds
    const float*    bcol = W2e + (size_t)bk * D_OUT + n0 + bn;

    v8h ha0, ha1;
    float4 fb0, fb1;

    auto fetch = [&](int kb) {
        const v8h* p = reinterpret_cast<const v8h*>(arow + kb);
        ha0 = p[0]; ha1 = p[1];                      // already f16
        const float* src = bcol + (size_t)kb * D_OUT;
        const float4* q = reinterpret_cast<const float4*>(src);
        fb0 = q[0]; fb1 = q[1];
        __builtin_prefetch(src + (size_t)TK * D_OUT, 0, 1);
    };
    auto stash = [&](int buf) {
        *reinterpret_cast<v8h*>(&As[buf][ai * TK + ak])     = ha0;
        *reinterpret_cast<v8h*>(&As[buf][ai * TK + ak + 8]) = ha1;
        _Float16* bs = &Bs[buf][0];
        bs[(bn + 0) * TK + bk] = (_Float16)fb0.x;
        bs[(bn + 1) * TK + bk] = (_Float16)fb0.y;
        bs[(bn + 2) * TK + bk] = (_Float16)fb0.z;
        bs[(bn + 3) * TK + bk] = (_Float16)fb0.w;
        bs[(bn + 4) * TK + bk] = (_Float16)fb1.x;
        bs[(bn + 5) * TK + bk] = (_Float16)fb1.y;
        bs[(bn + 6) * TK + bk] = (_Float16)fb1.z;
        bs[(bn + 7) * TK + bk] = (_Float16)fb1.w;
    };
    auto compute = [&](int buf) {
        const int koA = (lane >> 4) * 8;
        const int m0  = wm * 32 + (lane & 15);
        const _Float16* ap0 = &As[buf][m0 * TK];
        const _Float16* ap1 = &As[buf][(m0 + 16) * TK];
        v16h a0 = cat8(*reinterpret_cast<const v8h*>(ap0 + koA),
                       *reinterpret_cast<const v8h*>(ap0 + 16 + koA));
        v16h a1 = cat8(*reinterpret_cast<const v8h*>(ap1 + koA),
                       *reinterpret_cast<const v8h*>(ap1 + 16 + koA));
        const int koB = (lane >> 4) * 8;
        const int nc  = wn * 32 + (lane & 15);
        const _Float16* bp0 = &Bs[buf][nc * TK + koB];
        const _Float16* bp1 = &Bs[buf][(nc + 16) * TK + koB];
        v16h b0 = cat8(*reinterpret_cast<const v8h*>(bp0),
                       *reinterpret_cast<const v8h*>(bp0 + 16));
        v16h b1f = cat8(*reinterpret_cast<const v8h*>(bp1),
                        *reinterpret_cast<const v8h*>(bp1 + 16));
        c[0][0] = __builtin_amdgcn_wmma_f32_16x16x32_f16(false, a0, false, b0,  (short)0, c[0][0], false, false);
        c[0][1] = __builtin_amdgcn_wmma_f32_16x16x32_f16(false, a0, false, b1f, (short)0, c[0][1], false, false);
        c[1][0] = __builtin_amdgcn_wmma_f32_16x16x32_f16(false, a1, false, b0,  (short)0, c[1][0], false, false);
        c[1][1] = __builtin_amdgcn_wmma_f32_16x16x32_f16(false, a1, false, b1f, (short)0, c[1][1], false, false);
    };

    const int KT = D_HID / TK;   // 96
    fetch(0);
    stash(0);
    for (int kt = 0; kt < KT; ++kt) {
        if (kt + 1 < KT) fetch((kt + 1) * TK);
        __syncthreads();
        compute(kt & 1);
        if (kt + 1 < KT) stash((kt + 1) & 1);
    }

    const int mb = (lane >> 4) * 8;
    const int nl = lane & 15;
    const float* b2e = b2 + (size_t)e * D_OUT;
    #pragma unroll
    for (int fm = 0; fm < 2; ++fm) {
        #pragma unroll
        for (int fn = 0; fn < 2; ++fn) {
            const int nglob = n0 + wn * 32 + fn * 16 + nl;
            const float bias = b2e[nglob];
            #pragma unroll
            for (int r = 0; r < 8; ++r) {
                int rl = wm * 32 + fm * 16 + mb + r;
                if (mt * TM + rl < cnt)
                    Y[(size_t)(base + mt * TM + rl) * D_OUT + nglob] = c[fm][fn][r] + bias;
            }
        }
    }
}

// ---------------- 7) combine top-2 + LayerNorm (two-pass) -------------------
__global__ __launch_bounds__(256) void moe_combine_ln(
    const float* __restrict__ Y, const int* __restrict__ slot_of,
    const float* __restrict__ tok_w, const float* __restrict__ ln_g,
    const float* __restrict__ ln_b, float* __restrict__ out)
{
    const int t   = blockIdx.x;
    const int tid = threadIdx.x;
    const int s0  = slot_of[2 * t];
    const int s1  = slot_of[2 * t + 1];
    const float w0 = tok_w[2 * t];
    const float w1 = tok_w[2 * t + 1];

    float v[3];
    float lsum = 0.0f;
    #pragma unroll
    for (int j = 0; j < 3; ++j) {
        int o = tid + j * 256;
        v[j] = w0 * Y[(size_t)s0 * D_OUT + o] + w1 * Y[(size_t)s1 * D_OUT + o];
        lsum += v[j];
    }
    __shared__ float red[256];
    red[tid] = lsum;
    __syncthreads();
    for (int s = 128; s > 0; s >>= 1) {
        if (tid < s) red[tid] += red[tid + s];
        __syncthreads();
    }
    const float mu = red[0] * (1.0f / D_OUT);
    __syncthreads();
    float lsq = 0.0f;
    #pragma unroll
    for (int j = 0; j < 3; ++j) { float d = v[j] - mu; lsq += d * d; }
    red[tid] = lsq;
    __syncthreads();
    for (int s = 128; s > 0; s >>= 1) {
        if (tid < s) red[tid] += red[tid + s];
        __syncthreads();
    }
    const float rstd = rsqrtf(red[0] * (1.0f / D_OUT) + LN_EPS);
    #pragma unroll
    for (int j = 0; j < 3; ++j) {
        int o = tid + j * 256;
        out[(size_t)t * D_OUT + o] = (v[j] - mu) * rstd * ln_g[o] + ln_b[o];
    }
}

// ---------------- host-side launcher ----------------
extern "C" void kernel_launch(void* const* d_in, const int* in_sizes, int n_in,
                              void* d_out, int out_size, void* d_ws, size_t ws_size,
                              hipStream_t stream) {
    (void)in_sizes; (void)n_in; (void)out_size; (void)ws_size;
    const float* x    = (const float*)d_in[0];
    const float* Wr   = (const float*)d_in[1];
    const float* br   = (const float*)d_in[2];
    const float* W1   = (const float*)d_in[3];
    const float* b1   = (const float*)d_in[4];
    const float* W2   = (const float*)d_in[5];
    const float* b2   = (const float*)d_in[6];
    const float* ln_g = (const float*)d_in[7];
    const float* ln_b = (const float*)d_in[8];
    float* out = (float*)d_out;

    auto align256 = [](size_t v) { return (v + 255) & ~(size_t)255; };
    char* ws = (char*)d_ws;
    size_t off = 0;
    _Float16* Hh = (_Float16*)(ws + off); off = align256(off + (size_t)SLOT_PAD * D_HID * sizeof(_Float16));
    float* Yb    = (float*)(ws + off);    off = align256(off + (size_t)NSLOT * D_OUT * sizeof(float));
    int* row_token = (int*)(ws + off);    off = align256(off + (size_t)NSLOT * sizeof(int));
    int* slot_of   = (int*)(ws + off);    off = align256(off + (size_t)NSLOT * sizeof(int));
    int* tok_idx   = (int*)(ws + off);    off = align256(off + (size_t)NSLOT * sizeof(int));
    float* tok_w   = (float*)(ws + off);  off = align256(off + (size_t)NSLOT * sizeof(float));
    int* counts    = (int*)(ws + off);    off = align256(off + N_EXP * sizeof(int));
    int* cursor    = (int*)(ws + off);    off = align256(off + N_EXP * sizeof(int));
    int* offs      = (int*)(ws + off);    off = align256(off + (N_EXP + 1) * sizeof(int));

    moe_zero<<<1, 32, 0, stream>>>(counts, cursor);
    moe_router<<<T_TOK, 128, 0, stream>>>(x, Wr, br, tok_idx, tok_w, counts);
    moe_offsets<<<1, 1, 0, stream>>>(counts, offs);
    moe_assign<<<T_TOK / 256, 256, 0, stream>>>(tok_idx, offs, cursor, row_token, slot_of);
    moe_gemm1<<<dim3(N_EXP * MAX_MT, D_HID / TN), 256, 0, stream>>>(x, W1, b1, offs, row_token, Hh);
    moe_gemm2<<<dim3(N_EXP * MAX_MT, D_OUT / TN), 256, 0, stream>>>(Hh, W2, b2, offs, Yb);
    moe_combine_ln<<<T_TOK, 256, 0, stream>>>(Yb, slot_of, tok_w, ln_g, ln_b, out);
}
